// Decoder_62328565399589
// MI455X (gfx1250) — compile-verified, matrix operand order checked by reading, not run on hardware
//
#include <hip/hip_runtime.h>

#define V_ 50257
#define E_ 256
#define H_ 512
#define B_ 32
#define T_ 31
#define G_ 1536          // 3H
#define LDK_ 1024        // A-matrix K (two H halves)
#define LW_ 1280         // linear_w row length (E + 2H)

typedef __attribute__((ext_vector_type(16))) __bf16 v16bf;
typedef __attribute__((ext_vector_type(8)))  __bf16 v8bf;
typedef __attribute__((ext_vector_type(8)))  float  v8f;

__device__ __forceinline__ __bf16 f2bf(float f) {
  unsigned u = __builtin_bit_cast(unsigned, f);
  unsigned r = u + 0x7FFFu + ((u >> 16) & 1u);   // round-to-nearest-even
  unsigned short s = (unsigned short)(r >> 16);
  return __builtin_bit_cast(__bf16, s);
}
__device__ __forceinline__ unsigned short bf2us(__bf16 b) {
  return __builtin_bit_cast(unsigned short, b);
}

// 16x16x32 bf16 A/B fragment from a row-major matrix stored as ushort-bf16.
// p already points at element [row(lane)][kb(lane)]; fragment covers k=kc..kc+31.
__device__ __forceinline__ v16bf frag_bf16(const unsigned short* p, int kc) {
  v8bf lo = *(const v8bf*)(p + kc);        // K  kb..kb+7      (16B aligned)
  v8bf hi = *(const v8bf*)(p + kc + 16);   // K  kb+16..kb+23
  return __builtin_shufflevector(lo, hi, 0,1,2,3,4,5,6,7,8,9,10,11,12,13,14,15);
}
// Fallback fragment: convert fp32 weights to bf16 in-register (used only when
// the workspace is too small to hold the pre-converted weights).
__device__ __forceinline__ v16bf frag_f32(const float* p, int kc) {
  v16bf r;
#pragma unroll
  for (int j = 0; j < 8; ++j) {
    r[j]     = f2bf(p[kc + j]);
    r[j + 8] = f2bf(p[kc + j + 16]);
  }
  return r;
}
__device__ __forceinline__ v8f wmma_bf16(v16bf a, v16bf b, v8f c) {
  return __builtin_amdgcn_wmma_f32_16x16x32_bf16(false, a, false, b, (short)0, c,
                                                 false, false);
}
__device__ __forceinline__ float sigmf(float x) { return 1.f / (1.f + __expf(-x)); }

// ---------------------------------------------------------------- prep kernels
__global__ void k_cvec(const float* __restrict__ emb, const float* __restrict__ lw,
                       float* __restrict__ c_vec) {
  __shared__ float es[E_];
  for (int i = threadIdx.x; i < E_; i += blockDim.x) es[i] = emb[2 * E_ + i];
  __syncthreads();
  int v = blockIdx.x * blockDim.x + threadIdx.x;
  if (v >= V_) return;
  const float* row = lw + (long)v * LW_;
  float s = 0.f;
#pragma unroll 8
  for (int e = 0; e < E_; ++e) s += es[e] * row[e];
  c_vec[v] = s;
}

__global__ void k_ge0(const float* __restrict__ emb, const float* __restrict__ w_ih0,
                      const float* __restrict__ b_ih0, float* __restrict__ ge0) {
  __shared__ float es[E_];
  for (int i = threadIdx.x; i < E_; i += blockDim.x) es[i] = emb[2 * E_ + i];
  __syncthreads();
  int j = blockIdx.x * blockDim.x + threadIdx.x;
  if (j >= G_) return;
  const float* row = w_ih0 + (long)j * (E_ + H_);
  float s = b_ih0[j];
#pragma unroll 8
  for (int e = 0; e < E_; ++e) s += es[e] * row[e];
  ge0[j] = s;
}

__global__ void k_wcvt(const float* __restrict__ w_ih0, const float* __restrict__ w_hh0,
                       const float* __restrict__ w_ih1, const float* __restrict__ w_hh1,
                       unsigned short* __restrict__ Wih0h, unsigned short* __restrict__ Whh0,
                       unsigned short* __restrict__ Wih1,  unsigned short* __restrict__ Whh1) {
  int id = blockIdx.x * blockDim.x + threadIdx.x;
  const int n = G_ * H_;
  if (id < n) {
    int j = id / H_, k = id - j * H_;
    Wih0h[id] = bf2us(f2bf(w_ih0[(long)j * (E_ + H_) + E_ + k]));
  } else if (id < 2 * n) {
    Whh0[id - n] = bf2us(f2bf(w_hh0[id - n]));
  } else if (id < 3 * n) {
    Wih1[id - 2 * n] = bf2us(f2bf(w_ih1[id - 2 * n]));
  } else if (id < 4 * n) {
    Whh1[id - 3 * n] = bf2us(f2bf(w_hh1[id - 3 * n]));
  }
}

// Stream-convert linear_w[:, 256:1280] to bf16 once (read 206MB + write 103MB,
// pure bandwidth). Each thread handles 8 consecutive k of one vocab row so the
// compiler emits b128 load / b128 store.
__global__ void k_lwcvt(const float* __restrict__ lw, unsigned short* __restrict__ Wb) {
  long id = (long)blockIdx.x * blockDim.x + threadIdx.x;
  if (id >= (long)V_ * (LDK_ / 8)) return;
  int v = (int)(id >> 7);
  int k8 = ((int)id & 127) << 3;
  const float* src = lw + (long)v * LW_ + E_ + k8;
  unsigned short* dst = Wb + (long)v * LDK_ + k8;
#pragma unroll
  for (int j = 0; j < 8; ++j) dst[j] = bf2us(f2bf(src[j]));
}

// ------------------------------------------------- sequential GRU (1 workgroup)
__device__ void gate_phase(const unsigned short* A1, const unsigned short* W1,
                           const float* bias1,
                           const unsigned short* A2, const unsigned short* W2,
                           const float* bias2,
                           float* gi, float* gh, int tid) {
  int wv = tid >> 5, lane = tid & 31;
  int r = lane & 15;
  int kb = (lane & 16) ? 8 : 0;
  for (int task = wv; task < 384; task += 16) {
    int which = task / 192;
    int rem = task - which * 192;
    int mt = rem / 96;
    int nt = rem - mt * 96;
    const unsigned short* A = which ? A2 : A1;
    const unsigned short* W = which ? W2 : W1;
    const float* bias = which ? bias2 : bias1;
    float* out = which ? gh : gi;
    int m0 = mt * 16, n0 = nt * 16;
    const unsigned short* pa = A + (m0 + r) * H_ + kb;
    const unsigned short* pw = W + (n0 + r) * H_ + kb;
    float bv = bias[n0 + r];
    v8f acc;
#pragma unroll
    for (int i = 0; i < 8; ++i) acc[i] = bv;
#pragma unroll 4
    for (int kc = 0; kc < H_; kc += 32)
      acc = wmma_bf16(frag_bf16(pa, kc), frag_bf16(pw, kc), acc);
    int rb = (lane & 16) ? 8 : 0;
    int col = n0 + r;
#pragma unroll
    for (int i = 0; i < 8; ++i) out[(m0 + rb + i) * G_ + col] = acc[i];
  }
}

__device__ void gates_phase(const float* gi, const float* gh, float* hf,
                            unsigned short* hb, unsigned short* Aconc, int t, int tid) {
  for (int idx = tid; idx < B_ * H_; idx += 512) {
    int b = idx >> 9, hh = idx & (H_ - 1);
    const float* gib = gi + b * G_;
    const float* ghb = gh + b * G_;
    float rr = sigmf(gib[hh] + ghb[hh]);
    float zz = sigmf(gib[H_ + hh] + ghb[H_ + hh]);
    float nn = tanhf(gib[2 * H_ + hh] + rr * ghb[2 * H_ + hh]);
    float hv = (1.f - zz) * nn + zz * hf[idx];
    hf[idx] = hv;
    unsigned short u = bf2us(f2bf(hv));
    hb[idx] = u;
    if (Aconc) {                       // layer-1: publish h1_t into the A matrix
      int m = b * T_ + t;
      Aconc[(long)m * LDK_ + H_ + hh] = u;                   // "new" half, step t
      if (t + 1 < T_) Aconc[(long)(m + 1) * LDK_ + hh] = u;  // "prev" half, t+1
    }
  }
}

__global__ __launch_bounds__(512) void k_recur(
    const float* __restrict__ hiddens,
    const float* __restrict__ b_hh0, const float* __restrict__ b_ih1,
    const float* __restrict__ b_hh1, const float* __restrict__ ge0,
    const unsigned short* __restrict__ Wih0h, const unsigned short* __restrict__ Whh0,
    const unsigned short* __restrict__ Wih1,  const unsigned short* __restrict__ Whh1,
    unsigned short* __restrict__ Aconc,
    float* h0f, float* h1f, unsigned short* h0b, unsigned short* h1b,
    float* gi, float* gh) {
  int tid = threadIdx.x;
  for (int idx = tid; idx < B_ * H_; idx += 512) {
    float hv = hiddens[idx];
    h0f[idx] = hv; h1f[idx] = hv;
    unsigned short u = bf2us(f2bf(hv));
    h0b[idx] = u; h1b[idx] = u;
    int b = idx >> 9, hh = idx & (H_ - 1);
    Aconc[(long)(b * T_) * LDK_ + hh] = u;   // h1_{-1} is "prev" half of step 0
  }
  __threadfence(); __syncthreads();
  for (int t = 0; t < T_; ++t) {
    gate_phase(h1b, Wih0h, ge0, h0b, Whh0, b_hh0, gi, gh, tid);
    __threadfence(); __syncthreads();
    gates_phase(gi, gh, h0f, h0b, nullptr, t, tid);
    __threadfence(); __syncthreads();
    gate_phase(h0b, Wih1, b_ih1, h1b, Whh1, b_hh1, gi, gh, tid);
    __threadfence(); __syncthreads();
    gates_phase(gi, gh, h1f, h1b, Aconc, t, tid);
    __threadfence(); __syncthreads();
  }
}

// --------------------------------------- batched vocab projection (the 102 GFLOP)
// out[m, v] = c_vec[v] + A[m,:] @ W[v,:],  m = b*31+t  (M=992, K=1024, N=50257)
// Block: 256 threads = 8 waves; each wave owns a 32(M) x 64(N) tile:
// 2 A-frags + 4 B-frags -> 8 WMMA per k-step (12 b128 loads : 8 wmma).
template <bool PRE>
__global__ __launch_bounds__(256) void k_score(const float* __restrict__ lw,
                                               const unsigned short* __restrict__ Wb,
                                               const unsigned short* __restrict__ Aconc,
                                               const float* __restrict__ c_vec,
                                               float* __restrict__ out) {
  int lane = threadIdx.x & 31;
  int wv = threadIdx.x >> 5;
  int m0 = blockIdx.y * 32;
  int n0 = blockIdx.x * 512 + wv * 64;
  int r = lane & 15;
  int kb = (lane & 16) ? 8 : 0;

  const unsigned short* pa0 = Aconc + (long)(m0 + r) * LDK_ + kb;
  const unsigned short* pa1 = pa0 + 16 * LDK_;

  int vcol[4];
  const unsigned short* pwb[4];
  const float* pwf[4];
  v8f acc0[4], acc1[4];
#pragma unroll
  for (int ni = 0; ni < 4; ++ni) {
    vcol[ni] = n0 + ni * 16 + r;
    int c = vcol[ni] < V_ ? vcol[ni] : V_ - 1;
    pwb[ni] = Wb + (long)c * LDK_ + kb;
    pwf[ni] = lw + (long)c * LW_ + E_ + kb;
    float cv = (vcol[ni] < V_) ? c_vec[vcol[ni]] : 0.f;
#pragma unroll
    for (int i = 0; i < 8; ++i) { acc0[ni][i] = cv; acc1[ni][i] = cv; }
  }

  for (int kc = 0; kc < LDK_; kc += 32) {
    if (PRE) {
      __builtin_prefetch(pwb[0] + kc + 128, 0, 3);   // global_prefetch_b8
      __builtin_prefetch(pwb[2] + kc + 128, 0, 3);
    } else {
      __builtin_prefetch(pwf[0] + kc + 64, 0, 3);
      __builtin_prefetch(pwf[2] + kc + 64, 0, 3);
    }
    v16bf a0 = frag_bf16(pa0, kc);
    v16bf a1 = frag_bf16(pa1, kc);
#pragma unroll
    for (int ni = 0; ni < 4; ++ni) {
      v16bf b = PRE ? frag_bf16(pwb[ni], kc) : frag_f32(pwf[ni], kc);
      acc0[ni] = wmma_bf16(a0, b, acc0[ni]);
      acc1[ni] = wmma_bf16(a1, b, acc1[ni]);
    }
  }

  int rb = (lane & 16) ? 8 : 0;
#pragma unroll
  for (int ni = 0; ni < 4; ++ni) {
    if (vcol[ni] < V_) {
#pragma unroll
      for (int i = 0; i < 8; ++i) {
        out[(long)(m0 + rb + i) * V_ + vcol[ni]]      = acc0[ni][i];
        out[(long)(m0 + 16 + rb + i) * V_ + vcol[ni]] = acc1[ni][i];
      }
    }
  }
}

// --------------------------------------------------------------------- launch
extern "C" void kernel_launch(void* const* d_in, const int* in_sizes, int n_in,
                              void* d_out, int out_size, void* d_ws, size_t ws_size,
                              hipStream_t stream) {
  const float* hiddens = (const float*)d_in[0];
  const float* emb     = (const float*)d_in[1];
  const float* w_ih0   = (const float*)d_in[2];
  const float* w_hh0   = (const float*)d_in[3];
  const float* b_ih0   = (const float*)d_in[4];
  const float* b_hh0   = (const float*)d_in[5];
  const float* w_ih1   = (const float*)d_in[6];
  const float* w_hh1   = (const float*)d_in[7];
  const float* b_ih1   = (const float*)d_in[8];
  const float* b_hh1   = (const float*)d_in[9];
  const float* lw      = (const float*)d_in[10];
  float* out = (float*)d_out;

  char* p = (char*)d_ws;
  auto alloc = [&](size_t bytes) {
    char* r = p;
    p += (bytes + 255) & ~(size_t)255;
    return r;
  };
  float* c_vec = (float*)alloc((size_t)V_ * 4);
  float* ge0   = (float*)alloc((size_t)G_ * 4);
  unsigned short* Wih0h = (unsigned short*)alloc((size_t)G_ * H_ * 2);
  unsigned short* Whh0  = (unsigned short*)alloc((size_t)G_ * H_ * 2);
  unsigned short* Wih1  = (unsigned short*)alloc((size_t)G_ * H_ * 2);
  unsigned short* Whh1  = (unsigned short*)alloc((size_t)G_ * H_ * 2);
  unsigned short* Aconc = (unsigned short*)alloc((size_t)B_ * T_ * LDK_ * 2);
  float* h0f = (float*)alloc((size_t)B_ * H_ * 4);
  float* h1f = (float*)alloc((size_t)B_ * H_ * 4);
  unsigned short* h0b = (unsigned short*)alloc((size_t)B_ * H_ * 2);
  unsigned short* h1b = (unsigned short*)alloc((size_t)B_ * H_ * 2);
  float* gi = (float*)alloc((size_t)B_ * G_ * 4);
  float* gh = (float*)alloc((size_t)B_ * G_ * 4);

  // Pre-converted bf16 copy of linear_w[:, 256:1280] if workspace allows
  // (103 MB; decision depends only on ws_size -> deterministic).
  const size_t wb_bytes = (size_t)V_ * LDK_ * 2;
  size_t used = (size_t)(p - (char*)d_ws);
  bool pre = (used + wb_bytes) <= ws_size;
  unsigned short* Wb = pre ? (unsigned short*)alloc(wb_bytes) : (unsigned short*)0;

  k_cvec<<<(V_ + 255) / 256, 256, 0, stream>>>(emb, lw, c_vec);
  k_ge0<<<(G_ + 255) / 256, 256, 0, stream>>>(emb, w_ih0, b_ih0, ge0);
  k_wcvt<<<(4 * G_ * H_ + 255) / 256, 256, 0, stream>>>(
      w_ih0, w_hh0, w_ih1, w_hh1, Wih0h, Whh0, Wih1, Whh1);
  if (pre) {
    long n8 = (long)V_ * (LDK_ / 8);
    k_lwcvt<<<(unsigned)((n8 + 255) / 256), 256, 0, stream>>>(lw, Wb);
  }
  k_recur<<<1, 512, 0, stream>>>(hiddens, b_hh0, b_ih1, b_hh1, ge0,
                                 Wih0h, Whh0, Wih1, Whh1, Aconc,
                                 h0f, h1f, h0b, h1b, gi, gh);
  dim3 grid((V_ + 511) / 512, (B_ * T_) / 32);
  if (pre)
    k_score<true><<<grid, 256, 0, stream>>>(lw, Wb, Aconc, c_vec, out);
  else
    k_score<false><<<grid, 256, 0, stream>>>(lw, Wb, Aconc, c_vec, out);
}
